// MPNN_6167573037057
// MI455X (gfx1250) — matrix operand order repaired and checked by reading, not compiled, not run on hardware
//
#include <hip/hip_runtime.h>
#include <math.h>

// ---------------------------------------------------------------------------
// MPNN forward for MI455X (gfx1250, wave32, WMMA).
//
// Heavy path restructure: instead of materializing We (E,64,64)=1.31GB and
// streaming it from HBM 3x, we project per step  Y = x @ G  where
// G (64 x 1088) packs the 16 basis matrices T_k = W_edge[:,k].reshape(64,64)
// plus the bias matrix B. Y (40000 x 1088 fp32 = 174 MB) fits the 192MB L2,
// so the per-edge combine  msg = sum_k ef[e,k]*Y[src,k,:] + Y[src,16,:]
// is L2-resident. All GEMMs use v_wmma_f32_16x16x32_f16 (fp32 accumulate),
// 4 WMMA tiles (16x64 output) per wave sharing one A fragment. K is a
// template parameter (64/128/256 here) so the K loop fully unrolls and the
// scheduler can software-pipeline loads across chunks instead of emitting
// s_wait_loadcnt + hazard v_nops around every WMMA.
// ---------------------------------------------------------------------------

#define H_ 64
#define N_NODES_ 40000
#define N_EDGES_ 80000
#define N_GRAPHS_ 256
#define NODE_IN_ 128
#define EDGE_IN_ 16
#define READOUT_ 1024
#define NB_ 17                      // 16 basis + bias block
#define YW_ (NB_ * H_)              // 1088

typedef __attribute__((ext_vector_type(16))) _Float16 v16h;
typedef __attribute__((ext_vector_type(8)))  float    v8f;

static inline int cdiv(int a, int b) { return (a + b - 1) / b; }

// ------------------------- small utility kernels ---------------------------

__global__ void k_cvt_f16(const float* __restrict__ s, _Float16* __restrict__ d, int n) {
    int i = blockIdx.x * blockDim.x + threadIdx.x;
    if (i < n) d[i] = (_Float16)s[i];
}

// Build G as (1088 rows x 64 cols) row-major f16 (GEMM B operand, N x K):
//   G[c][h], c = k*64+o (k<16): W_edge[(h*64+o)*16 + k];  c = 16*64+o: b_edge[h*64+o]
__global__ void k_build_G(const float* __restrict__ We, const float* __restrict__ be,
                          _Float16* __restrict__ G) {
    int i = blockIdx.x * blockDim.x + threadIdx.x;   // 1088*64
    if (i >= YW_ * H_) return;
    int c = i >> 6;          // 0..1087
    int h = i & 63;
    int k = c >> 6;          // 0..16
    int o = c & 63;
    float v = (k < 16) ? We[(size_t)(h * 64 + o) * 16 + k] : be[h * 64 + o];
    G[i] = (_Float16)v;
}

__global__ void k_vec_add(const float* a, const float* b, float* c, int n) {
    int i = blockIdx.x * blockDim.x + threadIdx.x;
    if (i < n) c[i] = a[i] + b[i];
}

__global__ void k_zero(float* p, int n) {
    int i = blockIdx.x * blockDim.x + threadIdx.x;
    if (i < n) p[i] = 0.f;
}

// ----------------------------- WMMA GEMM -----------------------------------
// C[M,N] = act( [reluA](A[M,K] f32) @ Bw[N,K]^T (f16) + bias[N] )   (+= if accum)
// One wave computes a 16x64 tile (4 WMMAs sharing one A fragment).
// Requires N % 64 == 0, K % 32 == 0, M % 16 == 0 (true for every call here).
template<int K>
__global__ void k_wmma_gemm(const float* __restrict__ A, int lda,
                            const _Float16* __restrict__ Bw, int ldb,
                            float* __restrict__ C, int ldc,
                            const float* __restrict__ bias,
                            int M, int N,
                            int reluA, int accum, int act,
                            const float* __restrict__ prelu) {
    constexpr int NT = 4;                            // 16-col tiles per wave
    int lane = threadIdx.x & 31;
    int wave = threadIdx.x >> 5;
    int ngrp = N >> 6;                               // 64-col groups
    int total = (M >> 4) * ngrp;
    int tile = blockIdx.x * (blockDim.x >> 5) + wave;
    if (tile >= total) return;                       // uniform per wave: EXEC all-1
    int row0 = (tile / ngrp) << 4;
    int colg = (tile % ngrp) << 6;

    int r = lane & 15;                               // A row / B column within tile
    int half = lane >> 4;
    const float* ap = A + (size_t)(row0 + r) * lda;
    const _Float16* bp[NT];
#pragma unroll
    for (int t = 0; t < NT; ++t)
        bp[t] = Bw + (size_t)(colg + 16 * t + r) * ldb;

    int cn = lane & 15;
    int mb = (lane >> 4) << 3;
    v8f acc[NT];
    if (accum) {
#pragma unroll
        for (int t = 0; t < NT; ++t)
#pragma unroll
            for (int v = 0; v < 8; ++v)
                acc[t][v] = C[(size_t)(row0 + mb + v) * ldc + colg + 16 * t + cn];
    } else {
#pragma unroll
        for (int t = 0; t < NT; ++t)
            acc[t] = (v8f){0.f, 0.f, 0.f, 0.f, 0.f, 0.f, 0.f, 0.f};
    }

    // Fully unrolled over K (compile-time): lets the backend hoist the next
    // chunk's global_load_b128s above the current chunk's WMMAs and keep
    // fragments in distinct registers (no per-WMMA s_wait_loadcnt / v_nop).
#pragma unroll
    for (int k0 = 0; k0 < K; k0 += 32) {
        // B fragments (32x16): lane = column, half selects K 0..15/16..31,
        // 16 contiguous f16 per lane -> 2x global_load_b128 each.
        v16h bf[NT];
#pragma unroll
        for (int t = 0; t < NT; ++t)
#pragma unroll
            for (int j = 0; j < 8; ++j) {
                int kb = k0 + (half << 4) + (j << 1);
                bf[t][2 * j]     = bp[t][kb];
                bf[t][2 * j + 1] = bp[t][kb + 1];
            }

        // A fragment: lane = row, half selects K 0..15/16..31 (matches B), f32->f16 on load.
        // Per lane: one contiguous run of 16 f32 -> 4x global_load_b128 + cvt_pk.
        v16h af;
#pragma unroll
        for (int j = 0; j < 8; ++j) {
            int kb = k0 + (half << 4) + (j << 1);
            float a0 = ap[kb], a1 = ap[kb + 1];
            if (reluA) { a0 = fmaxf(a0, 0.f); a1 = fmaxf(a1, 0.f); }
            af[2 * j]     = (_Float16)a0;
            af[2 * j + 1] = (_Float16)a1;
        }

#pragma unroll
        for (int t = 0; t < NT; ++t)
            acc[t] = __builtin_amdgcn_wmma_f32_16x16x32_f16(
                false, af, false, bf[t], (short)0, acc[t], false, false);
    }

    float pw = (act == 2) ? *prelu : 0.f;
#pragma unroll
    for (int t = 0; t < NT; ++t) {
        float bv = bias ? bias[colg + 16 * t + cn] : 0.f;
#pragma unroll
        for (int v = 0; v < 8; ++v) {
            float x = acc[t][v] + bv;
            if (act == 1)      x = fmaxf(x, 0.f);
            else if (act == 2) x = (x >= 0.f) ? x : pw * x;
            C[(size_t)(row0 + mb + v) * ldc + colg + 16 * t + cn] = x;
        }
    }
}

// --------------------------- message passing -------------------------------

__global__ void k_init_agg(const float* __restrict__ bconv, float* __restrict__ agg, int total) {
    int i = blockIdx.x * blockDim.x + threadIdx.x;
    if (i < total) agg[i] = bconv[i & (H_ - 1)];
}

// 4 edges per 256-thread block: msg = sum_k ef[e,k]*Y[src, k*64+o] + Y[src, 16*64+o]
__global__ void k_edge_combine(const float* __restrict__ Y,
                               const float* __restrict__ ef,
                               const int* __restrict__ esrc,
                               const int* __restrict__ edst,
                               float* __restrict__ agg, int E) {
    __shared__ float sef[4][16];
    int le = threadIdx.x >> 6;               // 0..3
    int o  = threadIdx.x & 63;
    int e  = blockIdx.x * 4 + le;
    if (e >= E) return;
    if (o < 16) sef[le][o] = ef[(size_t)e * 16 + o];
    __syncthreads();
    const float* y = Y + (size_t)esrc[e] * YW_;
    float m = y[16 * 64 + o];
#pragma unroll
    for (int k = 0; k < 16; ++k) m += sef[le][k] * y[k * 64 + o];
    atomicAdd(&agg[(size_t)edst[e] * H_ + o], m);
}

__global__ void k_gru_update(const float* __restrict__ gi, const float* __restrict__ gh,
                             float* __restrict__ hid, int total) {
    int i = blockIdx.x * blockDim.x + threadIdx.x;
    if (i >= total) return;
    int n = i >> 6, c = i & 63;
    size_t b = (size_t)n * 192;
    float ir = gi[b + c],        hr = gh[b + c];
    float iz = gi[b + 64 + c],   hz = gh[b + 64 + c];
    float in = gi[b + 128 + c],  hn = gh[b + 128 + c];
    float rg = 1.f / (1.f + expf(-(ir + hr)));
    float z  = 1.f / (1.f + expf(-(iz + hz)));
    float nn = tanhf(in + rg * hn);
    float h  = hid[i];
    hid[i] = (1.f - z) * nn + z * h;
}

__global__ void k_build_feat(const float* __restrict__ x0, const float* __restrict__ hid,
                             float* __restrict__ feat, int total) {
    int i = blockIdx.x * blockDim.x + threadIdx.x;
    if (i >= total) return;
    int n = i >> 7, d = i & 127;
    feat[i] = (d < 64) ? x0[(size_t)n * 64 + d] : hid[(size_t)n * 64 + d - 64];
}

// ------------------------------- Set2Set -----------------------------------

__global__ void k_gstart(const int* __restrict__ gid, int* __restrict__ gs, int N) {
    int n = blockIdx.x * blockDim.x + threadIdx.x;
    if (n >= N) return;
    if (n == 0 || gid[n] != gid[n - 1]) gs[gid[n]] = n;
    if (n == 0) gs[N_GRAPHS_] = N;
}

__global__ void k_lstm_update(const float* __restrict__ gates,
                              float* __restrict__ h, float* __restrict__ c, int total) {
    int i = blockIdx.x * blockDim.x + threadIdx.x;
    if (i >= total) return;
    int b = i >> 7, d = i & 127;
    size_t base = (size_t)b * 512;
    float ig = 1.f / (1.f + expf(-gates[base + d]));
    float fg = 1.f / (1.f + expf(-gates[base + 128 + d]));
    float gg = tanhf(gates[base + 256 + d]);
    float og = 1.f / (1.f + expf(-gates[base + 384 + d]));
    float cv = fg * c[i] + ig * gg;
    c[i] = cv;
    h[i] = og * tanhf(cv);
}

// e[n] = dot(feat[n,:], h[gid[n],:]) over D=128; one wave per node
__global__ void k_attn(const float* __restrict__ feat, const float* __restrict__ h,
                       const int* __restrict__ gid, float* __restrict__ e, int N) {
    int w = (blockIdx.x * blockDim.x + threadIdx.x) >> 5;
    int lane = threadIdx.x & 31;
    if (w >= N) return;
    const float* f = feat + (size_t)w * 128;
    const float* q = h + (size_t)gid[w] * 128;
    float v = 0.f;
#pragma unroll
    for (int d = lane; d < 128; d += 32) v += f[d] * q[d];
#pragma unroll
    for (int off = 16; off; off >>= 1) v += __shfl_down(v, off);
    if (lane == 0) e[w] = v;
}

// One 128-thread block per graph: segment softmax + weighted readout,
// writes q_star[g] = concat(h[g], r[g])
__global__ void k_readout(const float* __restrict__ feat, const float* __restrict__ e,
                          const int* __restrict__ gs, const float* __restrict__ h,
                          float* __restrict__ q_star) {
    __shared__ float red[128];
    int g = blockIdx.x, tid = threadIdx.x;
    int s = gs[g], t = gs[g + 1];
    float mx = -INFINITY;
    for (int n = s + tid; n < t; n += 128) mx = fmaxf(mx, e[n]);
    red[tid] = mx; __syncthreads();
    for (int off = 64; off; off >>= 1) {
        if (tid < off) red[tid] = fmaxf(red[tid], red[tid + off]);
        __syncthreads();
    }
    mx = red[0]; __syncthreads();
    float sm = 0.f;
    for (int n = s + tid; n < t; n += 128) sm += expf(e[n] - mx);
    red[tid] = sm; __syncthreads();
    for (int off = 64; off; off >>= 1) {
        if (tid < off) red[tid] += red[tid + off];
        __syncthreads();
    }
    float denom = red[0]; __syncthreads();

    int d = tid;
    float r = 0.f;
    for (int base = s; base < t; base += 128) {
        int cnt = min(128, t - base);
        __syncthreads();
        if (tid < cnt) red[tid] = expf(e[base + tid] - mx) / denom;
        __syncthreads();
        for (int j = 0; j < cnt; ++j) r += red[j] * feat[(size_t)(base + j) * 128 + d];
    }
    q_star[(size_t)g * 256 + d]       = h[(size_t)g * 128 + d];
    q_star[(size_t)g * 256 + 128 + d] = r;
}

// ------------------------------ launcher -----------------------------------

extern "C" void kernel_launch(void* const* d_in, const int* in_sizes, int n_in,
                              void* d_out, int out_size, void* d_ws, size_t ws_size,
                              hipStream_t stream) {
    const float* node_feats = (const float*)d_in[0];
    const float* edge_feats = (const float*)d_in[1];
    const int*   edge_src   = (const int*)d_in[2];
    const int*   edge_dst   = (const int*)d_in[3];
    const int*   gid        = (const int*)d_in[4];
    const float* W_proj     = (const float*)d_in[5];
    const float* b_proj     = (const float*)d_in[6];
    const float* W_edge     = (const float*)d_in[7];
    const float* b_edge     = (const float*)d_in[8];
    const float* b_conv     = (const float*)d_in[9];
    const float* gru_W_ih   = (const float*)d_in[10];
    const float* gru_W_hh   = (const float*)d_in[11];
    const float* gru_b_ih   = (const float*)d_in[12];
    const float* gru_b_hh   = (const float*)d_in[13];
    const float* lstm_W_ih  = (const float*)d_in[14];
    const float* lstm_W_hh  = (const float*)d_in[15];
    const float* lstm_b_ih  = (const float*)d_in[16];
    const float* lstm_b_hh  = (const float*)d_in[17];
    const float* W_sp       = (const float*)d_in[18];
    const float* b_sp       = (const float*)d_in[19];
    const float* prelu_w    = (const float*)d_in[20];
    float* out = (float*)d_out;

    // workspace carve-out
    char* w = (char*)d_ws;
    auto alloc = [&](size_t bytes) { void* p = (void*)w; w += (bytes + 255) & ~(size_t)255; return p; };
    _Float16* Wproj_h = (_Float16*)alloc((size_t)64 * 128 * 2);
    _Float16* G_h     = (_Float16*)alloc((size_t)YW_ * 64 * 2);
    _Float16* gWih_h  = (_Float16*)alloc((size_t)192 * 64 * 2);
    _Float16* gWhh_h  = (_Float16*)alloc((size_t)192 * 64 * 2);
    _Float16* Wsp_h   = (_Float16*)alloc((size_t)1024 * 256 * 2);
    _Float16* lWih_h  = (_Float16*)alloc((size_t)512 * 256 * 2);
    _Float16* lWhh_h  = (_Float16*)alloc((size_t)512 * 128 * 2);
    float* lb     = (float*)alloc((size_t)512 * 4);
    float* x0     = (float*)alloc((size_t)N_NODES_ * 64 * 4);
    float* hid    = (float*)alloc((size_t)N_NODES_ * 64 * 4);
    float* agg    = (float*)alloc((size_t)N_NODES_ * 64 * 4);
    float* feat   = (float*)alloc((size_t)N_NODES_ * 128 * 4);
    float* ebuf   = (float*)alloc((size_t)N_NODES_ * 4);
    int*   gs     = (int*)alloc((size_t)(N_GRAPHS_ + 1) * 4);
    float* gates  = (float*)alloc((size_t)256 * 512 * 4);
    float* h_l    = (float*)alloc((size_t)256 * 128 * 4);
    float* c_l    = (float*)alloc((size_t)256 * 128 * 4);
    float* q_star = (float*)alloc((size_t)256 * 256 * 4);
    float* Y      = (float*)alloc((size_t)N_NODES_ * YW_ * 4);   // 174 MB, L2-resident
    float* gi = Y;                                   // Y dead by GRU time: alias
    float* gh = Y + (size_t)N_NODES_ * 192;

    const int T = 256;
    // --- weight prep (f16 conversion + G packing), cheap and deterministic ---
    k_cvt_f16<<<cdiv(64 * 128, T), T, 0, stream>>>(W_proj, Wproj_h, 64 * 128);
    k_build_G<<<cdiv(YW_ * 64, T), T, 0, stream>>>(W_edge, b_edge, G_h);
    k_cvt_f16<<<cdiv(192 * 64, T), T, 0, stream>>>(gru_W_ih, gWih_h, 192 * 64);
    k_cvt_f16<<<cdiv(192 * 64, T), T, 0, stream>>>(gru_W_hh, gWhh_h, 192 * 64);
    k_cvt_f16<<<cdiv(1024 * 256, T), T, 0, stream>>>(W_sp, Wsp_h, 1024 * 256);
    k_cvt_f16<<<cdiv(512 * 256, T), T, 0, stream>>>(lstm_W_ih, lWih_h, 512 * 256);
    k_cvt_f16<<<cdiv(512 * 128, T), T, 0, stream>>>(lstm_W_hh, lWhh_h, 512 * 128);
    k_vec_add<<<cdiv(512, T), T, 0, stream>>>(lstm_b_ih, lstm_b_hh, lb, 512);

    // K dispatched at compile time: 64 / 128 / 256
    auto gemm = [&](const float* A, int lda, const _Float16* B, int ldb, float* C, int ldc,
                    const float* bias, int M, int N, int K, int reluA, int accum, int act,
                    const float* pw) {
        int grid = cdiv((M / 16) * (N / 64), 8);     // wave computes 16x64
        switch (K) {
        case 64:
            k_wmma_gemm<64><<<grid, 256, 0, stream>>>(A, lda, B, ldb, C, ldc, bias,
                                                      M, N, reluA, accum, act, pw);
            break;
        case 128:
            k_wmma_gemm<128><<<grid, 256, 0, stream>>>(A, lda, B, ldb, C, ldc, bias,
                                                       M, N, reluA, accum, act, pw);
            break;
        default:
            k_wmma_gemm<256><<<grid, 256, 0, stream>>>(A, lda, B, ldb, C, ldc, bias,
                                                       M, N, reluA, accum, act, pw);
            break;
        }
    };

    // --- x0 = relu(node_feats @ W_proj^T + b_proj); hid = x0 ---
    gemm(node_feats, 128, Wproj_h, 128, x0, 64, b_proj, N_NODES_, 64, 128, 0, 0, 1, nullptr);
    hipMemcpyAsync(hid, x0, (size_t)N_NODES_ * 64 * 4, hipMemcpyDeviceToDevice, stream);

    // --- 3 message-passing steps ---
    for (int step = 0; step < 3; ++step) {
        // Y = x @ G  (basis projection; stays in L2)
        gemm(hid, 64, G_h, 64, Y, YW_, nullptr, N_NODES_, YW_, 64, 0, 0, 0, nullptr);
        k_init_agg<<<cdiv(N_NODES_ * 64, T), T, 0, stream>>>(b_conv, agg, N_NODES_ * 64);
        k_edge_combine<<<cdiv(N_EDGES_, 4), 256, 0, stream>>>(Y, edge_feats, edge_src,
                                                              edge_dst, agg, N_EDGES_);
        // GRU gates (gi/gh alias the now-dead Y region)
        gemm(agg, 64, gWih_h, 64, gi, 192, gru_b_ih, N_NODES_, 192, 64, 1, 0, 0, nullptr);
        gemm(hid, 64, gWhh_h, 64, gh, 192, gru_b_hh, N_NODES_, 192, 64, 0, 0, 0, nullptr);
        k_gru_update<<<cdiv(N_NODES_ * 64, T), T, 0, stream>>>(gi, gh, hid, N_NODES_ * 64);
    }

    // --- feat = concat(x0, hid) ---
    k_build_feat<<<cdiv(N_NODES_ * 128, T), T, 0, stream>>>(x0, hid, feat, N_NODES_ * 128);
    k_gstart<<<cdiv(N_NODES_, T), T, 0, stream>>>(gid, gs, N_NODES_);
    k_zero<<<cdiv(256 * 128, T), T, 0, stream>>>(h_l, 256 * 128);
    k_zero<<<cdiv(256 * 128, T), T, 0, stream>>>(c_l, 256 * 128);
    k_zero<<<cdiv(256 * 256, T), T, 0, stream>>>(q_star, 256 * 256);

    // --- 3 Set2Set steps ---
    for (int step = 0; step < 3; ++step) {
        gemm(q_star, 256, lWih_h, 256, gates, 512, lb, 256, 512, 256, 0, 0, 0, nullptr);
        gemm(h_l, 128, lWhh_h, 128, gates, 512, nullptr, 256, 512, 128, 0, 1, 0, nullptr);
        k_lstm_update<<<cdiv(256 * 128, T), T, 0, stream>>>(gates, h_l, c_l, 256 * 128);
        k_attn<<<cdiv(N_NODES_ * 32, T), T, 0, stream>>>(feat, h_l, gid, ebuf, N_NODES_);
        k_readout<<<N_GRAPHS_, 128, 0, stream>>>(feat, ebuf, gs, h_l, q_star);
    }

    // --- out = prelu(q_star @ W_sp^T + b_sp) ---
    gemm(q_star, 256, Wsp_h, 256, out, READOUT_, b_sp, 256, READOUT_, 256, 0, 0, 2, prelu_w);

    (void)in_sizes; (void)n_in; (void)out_size; (void)ws_size;
}
